// VRNN_10660108829073
// MI455X (gfx1250) — compile-verified, matrix operand order checked by reading, not run on hardware
//
#include <hip/hip_runtime.h>
#include <hip/hip_bf16.h>
#include <math.h>

// ---------------------------------------------------------------------------
// Types for CDNA5 WMMA (wave32, 16x16x32 bf16 -> f32)
// ---------------------------------------------------------------------------
typedef __bf16 bf16;
typedef bf16  v16bf __attribute__((ext_vector_type(16)));
typedef float v8f   __attribute__((ext_vector_type(8)));
typedef int   v4i   __attribute__((ext_vector_type(4)));

union FragBF { v16bf v; uint4 q[2]; };

// Problem constants
#define XD 256
#define HD 1024
#define ZD 128
#define SL 128
#define BT 256

// CDNA5 async global->LDS copy (ASYNCcnt-tracked DMA), if the toolchain has it
#if defined(__HIP_DEVICE_COMPILE__) && __has_builtin(__builtin_amdgcn_global_load_async_to_lds_b128)
#define HAVE_ASYNC_LDS 1
#else
#define HAVE_ASYNC_LDS 0
#endif

#if HAVE_ASYNC_LDS
// builtin signature: (int4 addrspace(1)* src, int4 addrspace(3)* dst, imm offset, imm cpol)
#define GAS1(p) ((__attribute__((address_space(1))) v4i*)(p))
#define LAS3(p) ((__attribute__((address_space(3))) v4i*)(p))
__device__ __forceinline__ void async_wait0() {
#if __has_builtin(__builtin_amdgcn_s_wait_asynccnt)
    __builtin_amdgcn_s_wait_asynccnt(0);
#else
    asm volatile("s_wait_asynccnt 0x0" ::: "memory");
#endif
}
#endif

// ---------------------------------------------------------------------------
// GEMM: C[M,N] = act( A[M,K](bf16) @ W[N,K]^T(bf16) + bias[N] )   (C is f32)
// Block = 128 threads (4 waves) -> 64x64 C tile, waves in 2x2 grid of 32x32.
// Double-buffered LDS; async (or pipelined) global->LDS staging; K step 32.
// ACT: 0=none 1=relu 2=sigmoid 3=softplus
// ---------------------------------------------------------------------------
template<int ACT>
__global__ __launch_bounds__(128)
void gemm_bf16(const bf16* __restrict__ A, const bf16* __restrict__ W,
               const float* __restrict__ bias, float* __restrict__ C,
               int M, int N, int K)
{
    __shared__ __align__(16) bf16 As[2][64 * 32];
    __shared__ __align__(16) bf16 Bs[2][64 * 32];

    const int tid  = threadIdx.x;
    const int wave = tid >> 5;
    const int lane = tid & 31;
    const int wr   = wave >> 1;       // M half of block tile
    const int wc   = wave & 1;        // N half of block tile
    const int m0   = blockIdx.y * 64;
    const int n0   = blockIdx.x * 64;
    const int r    = lane & 15;       // row within 16-row fragment
    const int kh   = lane >> 4;       // k-half select

    // Staging slots: thread covers uint4 slots {tid, tid+128} of each 64x32 tile
    const int srow = tid >> 2;        // rows 0..31 (and +32)
    const int sc8  = (tid & 3) * 8;   // bf16 col offset 0/8/16/24

    v8f acc[2][2];
    #pragma unroll
    for (int p = 0; p < 2; ++p)
        #pragma unroll
        for (int q = 0; q < 2; ++q)
            #pragma unroll
            for (int v = 0; v < 8; ++v) acc[p][q][v] = 0.0f;

    const int nk = K >> 5;

#if HAVE_ASYNC_LDS
    auto stage = [&](int k0, int buf) {
        __builtin_amdgcn_global_load_async_to_lds_b128(
            GAS1(&A[(size_t)(m0 + srow) * K + k0 + sc8]),
            LAS3(&As[buf][srow * 32 + sc8]), 0, 0);
        __builtin_amdgcn_global_load_async_to_lds_b128(
            GAS1(&A[(size_t)(m0 + srow + 32) * K + k0 + sc8]),
            LAS3(&As[buf][(srow + 32) * 32 + sc8]), 0, 0);
        __builtin_amdgcn_global_load_async_to_lds_b128(
            GAS1(&W[(size_t)(n0 + srow) * K + k0 + sc8]),
            LAS3(&Bs[buf][srow * 32 + sc8]), 0, 0);
        __builtin_amdgcn_global_load_async_to_lds_b128(
            GAS1(&W[(size_t)(n0 + srow + 32) * K + k0 + sc8]),
            LAS3(&Bs[buf][(srow + 32) * 32 + sc8]), 0, 0);
    };
    stage(0, 0);
    async_wait0();
    __syncthreads();
#else
    uint4 ra0, ra1, rb0, rb1;
    auto loadtile = [&](int k0) {   // batch all global loads (single wait later)
        ra0 = *(const uint4*)&A[(size_t)(m0 + srow) * K + k0 + sc8];
        ra1 = *(const uint4*)&A[(size_t)(m0 + srow + 32) * K + k0 + sc8];
        rb0 = *(const uint4*)&W[(size_t)(n0 + srow) * K + k0 + sc8];
        rb1 = *(const uint4*)&W[(size_t)(n0 + srow + 32) * K + k0 + sc8];
    };
    auto commit = [&](int buf) {
        *(uint4*)&As[buf][srow * 32 + sc8]        = ra0;
        *(uint4*)&As[buf][(srow + 32) * 32 + sc8] = ra1;
        *(uint4*)&Bs[buf][srow * 32 + sc8]        = rb0;
        *(uint4*)&Bs[buf][(srow + 32) * 32 + sc8] = rb1;
    };
    loadtile(0);
    commit(0);
    __syncthreads();
#endif

    for (int i = 0; i < nk; ++i) {
        const int cb = i & 1;
#if HAVE_ASYNC_LDS
        if (i + 1 < nk) stage((i + 1) << 5, cb ^ 1);   // DMA next tile into alt buf
#else
        if (i + 1 < nk) loadtile((i + 1) << 5);        // issue loads, no wait yet
#endif
        const bf16* Ab = As[cb];
        const bf16* Bb = Bs[cb];
        FragBF afr[2], bfr[2];
        #pragma unroll
        for (int p = 0; p < 2; ++p) {
            const int arow = 32 * wr + 16 * p + r;
            afr[p].q[0] = *(const uint4*)&Ab[arow * 32 + kh * 8];
            afr[p].q[1] = *(const uint4*)&Ab[arow * 32 + 16 + kh * 8];
            const int brow = 32 * wc + 16 * p + r;
            bfr[p].q[0] = *(const uint4*)&Bb[brow * 32 + kh * 8];
            bfr[p].q[1] = *(const uint4*)&Bb[brow * 32 + 16 + kh * 8];
        }
        #pragma unroll
        for (int p = 0; p < 2; ++p)
            #pragma unroll
            for (int q = 0; q < 2; ++q)
                acc[p][q] = __builtin_amdgcn_wmma_f32_16x16x32_bf16(
                    false, afr[p].v, false, bfr[q].v, (short)0, acc[p][q], false, false);
#if HAVE_ASYNC_LDS
        if (i + 1 < nk) async_wait0();                 // DMA landed in alt buf
#else
        if (i + 1 < nk) commit(cb ^ 1);                // single wait, then ds stores
#endif
        __syncthreads();
    }

    // Epilogue: C f32 frag layout -> col = lane%16, VGPR v -> row v + 8*(lane/16)
    const int rb_ = m0 + 32 * wr + 8 * kh;
    const int cb_ = n0 + 32 * wc + r;
    #pragma unroll
    for (int p = 0; p < 2; ++p)
        #pragma unroll
        for (int q = 0; q < 2; ++q) {
            const int col = cb_ + 16 * q;
            const float bv = bias ? bias[col] : 0.0f;
            const int rowb = rb_ + 16 * p;
            #pragma unroll
            for (int v = 0; v < 8; ++v) {
                float val = acc[p][q][v] + bv;
                if (ACT == 1)      val = fmaxf(val, 0.0f);
                else if (ACT == 2) val = 1.0f / (1.0f + __expf(-val));
                else if (ACT == 3) val = fmaxf(val, 0.0f) + log1pf(__expf(-fabsf(val)));
                C[(size_t)(rowb + v) * N + col] = val;
            }
        }
}

// ---------------------------------------------------------------------------
// Elementwise helpers
// ---------------------------------------------------------------------------
__global__ void k_cast(const float* __restrict__ s, bf16* __restrict__ d, int n)
{
    for (int i = blockIdx.x * blockDim.x + threadIdx.x; i < n;
         i += gridDim.x * blockDim.x)
        d[i] = (bf16)s[i];
}

__global__ void k_cast_rows(const float* __restrict__ s, int scols,
                            bf16* __restrict__ d, int dstride, int rows, int cols)
{
    int n = rows * cols;
    for (int i = blockIdx.x * blockDim.x + threadIdx.x; i < n;
         i += gridDim.x * blockDim.x) {
        int rr = i / cols, cc = i - rr * cols;
        d[(size_t)rr * dstride + cc] = (bf16)s[(size_t)rr * scols + cc];
    }
}

__global__ void k_copy_bf(const bf16* __restrict__ s, int sstride,
                          bf16* __restrict__ d, int dstride, int rows, int cols)
{
    int n = rows * cols;
    for (int i = blockIdx.x * blockDim.x + threadIdx.x; i < n;
         i += gridDim.x * blockDim.x) {
        int rr = i / cols, cc = i - rr * cols;
        d[(size_t)rr * dstride + cc] = s[(size_t)rr * sstride + cc];
    }
}

// z = eps * enc_std + enc_mean   (bf16 out, feeds phi_z GEMM)
__global__ void k_z(const float* __restrict__ eps_t, const float* __restrict__ em,
                    const float* __restrict__ es, bf16* __restrict__ zbf, int n)
{
    for (int i = blockIdx.x * blockDim.x + threadIdx.x; i < n;
         i += gridDim.x * blockDim.x)
        zbf[i] = (bf16)(eps_t[i] * es[i] + em[i]);
}

// GRU cell: updates h (f32, in place) + bf16 copies into concat buffers
__global__ void k_gru(const float* __restrict__ gi, const float* __restrict__ gh,
                      float* __restrict__ h, bf16* __restrict__ h_bf,
                      bf16* __restrict__ enc_in, bf16* __restrict__ dec_in)
{
    int i = blockIdx.x * blockDim.x + threadIdx.x;
    if (i >= BT * HD) return;
    int b = i >> 10, c = i & 1023;
    size_t gb = (size_t)b * 3 * HD;
    float ir = gi[gb + c], iz = gi[gb + HD + c], in_ = gi[gb + 2 * HD + c];
    float hr = gh[gb + c], hz = gh[gb + HD + c], hn  = gh[gb + 2 * HD + c];
    float rg = 1.0f / (1.0f + __expf(-(ir + hr)));
    float zg = 1.0f / (1.0f + __expf(-(iz + hz)));
    float ng = tanhf(in_ + rg * hn);
    float hv = (1.0f - zg) * ng + zg * h[i];
    h[i] = hv;
    bf16 hb = (bf16)hv;
    h_bf[i] = hb;
    enc_in[(size_t)b * 2048 + 1024 + c] = hb;
    dec_in[(size_t)b * 2048 + 1024 + c] = hb;
}

__global__ void k_init(float* __restrict__ h, bf16* __restrict__ h_bf,
                       bf16* __restrict__ enc_in, bf16* __restrict__ dec_in,
                       float* __restrict__ out01)
{
    int i = blockIdx.x * blockDim.x + threadIdx.x;
    if (i < 2) out01[i] = 0.0f;
    if (i < BT * HD) {
        h[i] = 0.0f;
        h_bf[i] = (bf16)0.0f;
        int b = i >> 10, c = i & 1023;
        enc_in[(size_t)b * 2048 + 1024 + c] = (bf16)0.0f;
        dec_in[(size_t)b * 2048 + 1024 + c] = (bf16)0.0f;
    }
}

// KLD per step, accumulated into out[0]
__global__ __launch_bounds__(1024)
void k_kld(const float* __restrict__ em, const float* __restrict__ es,
           const float* __restrict__ pm, const float* __restrict__ ps,
           float* __restrict__ out, int n)
{
    __shared__ float red[1024];
    float s = 0.0f;
    for (int i = threadIdx.x; i < n; i += 1024) {
        float e = es[i], p = ps[i], d = em[i] - pm[i];
        s += 2.0f * __logf(p) - 2.0f * __logf(e) + (e * e + d * d) / (p * p) - 1.0f;
    }
    red[threadIdx.x] = s;
    __syncthreads();
    for (int off = 512; off > 0; off >>= 1) {
        if (threadIdx.x < off) red[threadIdx.x] += red[threadIdx.x + off];
        __syncthreads();
    }
    if (threadIdx.x == 0) out[0] += 0.5f * red[0];
}

// NLL per step: sqrt(sum (x - dec_mean)^2), accumulated into out[1]
__global__ __launch_bounds__(1024)
void k_nll(const float* __restrict__ x_t, const float* __restrict__ dm,
           float* __restrict__ out, int n)
{
    __shared__ float red[1024];
    float s = 0.0f;
    for (int i = threadIdx.x; i < n; i += 1024) {
        float d = x_t[i] - dm[i];
        s += d * d;
    }
    red[threadIdx.x] = s;
    __syncthreads();
    for (int off = 512; off > 0; off >>= 1) {
        if (threadIdx.x < off) red[threadIdx.x] += red[threadIdx.x + off];
        __syncthreads();
    }
    if (threadIdx.x == 0) out[1] += sqrtf(red[0]);
}

// ---------------------------------------------------------------------------
// Host-side launcher
// ---------------------------------------------------------------------------
static void gemm_go(int act, const bf16* A, const bf16* W, const float* b,
                    float* C, int M, int N, int K, hipStream_t s)
{
    dim3 g(N / 64, M / 64), blk(128);
    switch (act) {
    case 0: gemm_bf16<0><<<g, blk, 0, s>>>(A, W, b, C, M, N, K); break;
    case 1: gemm_bf16<1><<<g, blk, 0, s>>>(A, W, b, C, M, N, K); break;
    case 2: gemm_bf16<2><<<g, blk, 0, s>>>(A, W, b, C, M, N, K); break;
    default: gemm_bf16<3><<<g, blk, 0, s>>>(A, W, b, C, M, N, K); break;
    }
}

extern "C" void kernel_launch(void* const* d_in, const int* in_sizes, int n_in,
                              void* d_out, int out_size, void* d_ws, size_t ws_size,
                              hipStream_t stream)
{
    (void)in_sizes; (void)n_in; (void)out_size; (void)ws_size;

    const float* x   = (const float*)d_in[0];   // [128,256,256]
    const float* eps = (const float*)d_in[1];   // [128,256,128]
    // Weights/biases (f32) in params dict order
    const float* W1  = (const float*)d_in[2];   const float* b1  = (const float*)d_in[3];
    const float* W2  = (const float*)d_in[4];   const float* b2  = (const float*)d_in[5];
    const float* Wz  = (const float*)d_in[6];   const float* bz  = (const float*)d_in[7];
    const float* We1 = (const float*)d_in[8];   const float* be1 = (const float*)d_in[9];
    const float* We2 = (const float*)d_in[10];  const float* be2 = (const float*)d_in[11];
    const float* Wem = (const float*)d_in[12];  const float* bem = (const float*)d_in[13];
    const float* Wes = (const float*)d_in[14];  const float* bes = (const float*)d_in[15];
    const float* Wp  = (const float*)d_in[16];  const float* bp  = (const float*)d_in[17];
    const float* Wpm = (const float*)d_in[18];  const float* bpm = (const float*)d_in[19];
    const float* Wps = (const float*)d_in[20];  const float* bps = (const float*)d_in[21];
    const float* Wd1 = (const float*)d_in[22];  const float* bd1 = (const float*)d_in[23];
    const float* Wd2 = (const float*)d_in[24];  const float* bd2 = (const float*)d_in[25];
    const float* Wdm = (const float*)d_in[26];  const float* bdm = (const float*)d_in[27];
    const float* Wds = (const float*)d_in[28];  const float* bds = (const float*)d_in[29];
    const float* Wih = (const float*)d_in[30];
    const float* Whh = (const float*)d_in[31];

    float* out = (float*)d_out;
    // d_out layout: [kld, nll, em(128*256*128), es(..), dm(128*256*256), ds(..)]
    const size_t OFF_EM = 2;
    const size_t OFF_ES = OFF_EM + (size_t)SL * BT * ZD;
    const size_t OFF_DM = OFF_ES + (size_t)SL * BT * ZD;
    const size_t OFF_DS = OFF_DM + (size_t)SL * BT * XD;

    // Bump allocator over d_ws
    char* wsb = (char*)d_ws;
    size_t cur = 0;
    auto alloc = [&](size_t bytes) -> void* {
        void* p = wsb + cur;
        cur += (bytes + 255) & ~(size_t)255;
        return p;
    };
    auto castN = [&](const float* s, bf16* d, size_t n) {
        int blocks = (int)((n + 255) / 256); if (blocks > 2048) blocks = 2048;
        k_cast<<<blocks, 256, 0, stream>>>(s, d, (int)n);
    };
    auto bfw = [&](const float* src, size_t n) -> bf16* {
        bf16* p = (bf16*)alloc(n * sizeof(bf16));
        castN(src, p, n);
        return p;
    };

    // --- weights to bf16 (L2-resident: ~38.5 MB total) ---
    bf16* W1b  = bfw(W1,  (size_t)HD * XD);
    bf16* W2b  = bfw(W2,  (size_t)HD * HD);
    bf16* Wzb  = bfw(Wz,  (size_t)HD * ZD);
    bf16* We1b = bfw(We1, (size_t)HD * 2 * HD);
    bf16* We2b = bfw(We2, (size_t)HD * HD);
    bf16* Wemb = bfw(Wem, (size_t)ZD * HD);
    bf16* Wesb = bfw(Wes, (size_t)ZD * HD);
    bf16* Wpb  = bfw(Wp,  (size_t)HD * HD);
    bf16* Wpmb = bfw(Wpm, (size_t)ZD * HD);
    bf16* Wpsb = bfw(Wps, (size_t)ZD * HD);
    bf16* Wd1b = bfw(Wd1, (size_t)HD * 2 * HD);
    bf16* Wd2b = bfw(Wd2, (size_t)HD * HD);
    bf16* Wdmb = bfw(Wdm, (size_t)XD * HD);
    bf16* Wdsb = bfw(Wds, (size_t)XD * HD);
    bf16* Wihb = bfw(Wih, (size_t)3 * HD * 2 * HD);
    bf16* Whhb = bfw(Whh, (size_t)3 * HD * HD);

    // --- precompute phi_x for all timesteps (M = 128*256 = 32768) ---
    const int MT = SL * BT;            // 32768
    bf16*  x_bf      = (bf16*)alloc((size_t)MT * XD * sizeof(bf16));
    bf16*  phi_x_all = (bf16*)alloc((size_t)MT * HD * sizeof(bf16));
    float* tmpBig    = (float*)alloc((size_t)4096 * HD * sizeof(float));
    bf16*  tmpBigBf  = (bf16*)alloc((size_t)4096 * HD * sizeof(bf16));
    float* tmpBig2   = (float*)alloc((size_t)4096 * HD * sizeof(float));
    castN(x, x_bf, (size_t)MT * XD);
    for (int c = 0; c < 8; ++c) {
        gemm_go(1, x_bf + (size_t)c * 4096 * XD, W1b, b1, tmpBig, 4096, HD, XD, stream);
        castN(tmpBig, tmpBigBf, (size_t)4096 * HD);
        gemm_go(1, tmpBigBf, W2b, b2, tmpBig2, 4096, HD, HD, stream);
        castN(tmpBig2, phi_x_all + (size_t)c * 4096 * HD, (size_t)4096 * HD);
    }

    // --- per-step buffers ---
    bf16*  enc_in = (bf16*)alloc((size_t)BT * 2 * HD * sizeof(bf16)); // [phi_x | h]
    bf16*  dec_in = (bf16*)alloc((size_t)BT * 2 * HD * sizeof(bf16)); // [phi_z | h]
    bf16*  xz_cat = (bf16*)alloc((size_t)BT * 2 * HD * sizeof(bf16)); // [phi_x | phi_z]
    float* h      = (float*)alloc((size_t)BT * HD * sizeof(float));
    bf16*  h_bf   = (bf16*)alloc((size_t)BT * HD * sizeof(bf16));
    float* tmpS   = (float*)alloc((size_t)BT * HD * sizeof(float));
    bf16*  tmpSb  = (bf16*)alloc((size_t)BT * HD * sizeof(bf16));
    float* tmpS2  = (float*)alloc((size_t)BT * HD * sizeof(float));
    bf16*  tmpS2b = (bf16*)alloc((size_t)BT * HD * sizeof(bf16));
    float* pm     = (float*)alloc((size_t)BT * ZD * sizeof(float));
    float* ps     = (float*)alloc((size_t)BT * ZD * sizeof(float));
    float* gi     = (float*)alloc((size_t)BT * 3 * HD * sizeof(float));
    float* gh     = (float*)alloc((size_t)BT * 3 * HD * sizeof(float));
    bf16*  z_bf   = (bf16*)alloc((size_t)BT * ZD * sizeof(bf16));

    k_init<<<(BT * HD + 255) / 256, 256, 0, stream>>>(h, h_bf, enc_in, dec_in, out);

    // --- sequential recurrence ---
    for (int t = 0; t < SL; ++t) {
        const bf16*  phx  = phi_x_all + (size_t)t * BT * HD;
        float*       em_t = out + OFF_EM + (size_t)t * BT * ZD;
        float*       es_t = out + OFF_ES + (size_t)t * BT * ZD;
        float*       dm_t = out + OFF_DM + (size_t)t * BT * XD;
        float*       ds_t = out + OFF_DS + (size_t)t * BT * XD;
        const float* x_t  = x + (size_t)t * BT * XD;
        const float* e_t  = eps + (size_t)t * BT * ZD;

        // phi_x into concat buffers
        k_copy_bf<<<512, 256, 0, stream>>>(phx, HD, enc_in, 2 * HD, BT, HD);
        k_copy_bf<<<512, 256, 0, stream>>>(phx, HD, xz_cat, 2 * HD, BT, HD);

        // prior = relu(h @ Wp^T + bp); heads
        gemm_go(1, h_bf, Wpb, bp, tmpS, BT, HD, HD, stream);
        castN(tmpS, tmpSb, (size_t)BT * HD);
        gemm_go(0, tmpSb, Wpmb, bpm, pm, BT, ZD, HD, stream);
        gemm_go(3, tmpSb, Wpsb, bps, ps, BT, ZD, HD, stream);

        // gh = h @ Whh^T (uses old h)
        gemm_go(0, h_bf, Whhb, nullptr, gh, BT, 3 * HD, HD, stream);

        // enc = relu(relu(enc_in @ We1^T + be1) @ We2^T + be2); heads -> d_out
        gemm_go(1, enc_in, We1b, be1, tmpS, BT, HD, 2 * HD, stream);
        castN(tmpS, tmpSb, (size_t)BT * HD);
        gemm_go(1, tmpSb, We2b, be2, tmpS2, BT, HD, HD, stream);
        castN(tmpS2, tmpS2b, (size_t)BT * HD);
        gemm_go(0, tmpS2b, Wemb, bem, em_t, BT, ZD, HD, stream);
        gemm_go(3, tmpS2b, Wesb, bes, es_t, BT, ZD, HD, stream);

        // z = eps*std + mean ; phi_z = relu(z @ Wz^T + bz)
        k_z<<<128, 256, 0, stream>>>(e_t, em_t, es_t, z_bf, BT * ZD);
        gemm_go(1, z_bf, Wzb, bz, tmpS, BT, HD, ZD, stream);
        k_cast_rows<<<512, 256, 0, stream>>>(tmpS, HD, dec_in, 2 * HD, BT, HD);
        k_cast_rows<<<512, 256, 0, stream>>>(tmpS, HD, xz_cat + HD, 2 * HD, BT, HD);

        // dec; heads -> d_out
        gemm_go(1, dec_in, Wd1b, bd1, tmpS, BT, HD, 2 * HD, stream);
        castN(tmpS, tmpSb, (size_t)BT * HD);
        gemm_go(1, tmpSb, Wd2b, bd2, tmpS2, BT, HD, HD, stream);
        castN(tmpS2, tmpS2b, (size_t)BT * HD);
        gemm_go(2, tmpS2b, Wdmb, bdm, dm_t, BT, XD, HD, stream);
        gemm_go(3, tmpS2b, Wdsb, bds, ds_t, BT, XD, HD, stream);

        // GRU: gi = [phi_x|phi_z] @ Wih^T ; h update
        gemm_go(0, xz_cat, Wihb, nullptr, gi, BT, 3 * HD, 2 * HD, stream);
        k_gru<<<(BT * HD + 255) / 256, 256, 0, stream>>>(gi, gh, h, h_bf, enc_in, dec_in);

        // scalars
        k_kld<<<1, 1024, 0, stream>>>(em_t, es_t, pm, ps, out, BT * ZD);
        k_nll<<<1, 1024, 0, stream>>>(x_t, dm_t, out, BT * XD);
    }
}
